// OssammaNERBlock_29944511988527
// MI455X (gfx1250) — matrix operand order repaired and checked by reading, not compiled, not run on hardware
//
#include <hip/hip_runtime.h>
#include <hip/hip_bf16.h>
#include <math.h>

// ---------------- problem constants ----------------
#define Dd   512
#define Tt   1024
#define Bb   8
#define Hh   8
#define HD   64
#define Ss   256
#define TDc  128
#define WIN  256
#define HID  768
#define Mm   (Bb*Tt)      // 8192 rows (b*T+t)
#define MH   (Mm*Hh)      // 65536 rows (b*T+t)*H+h

typedef _Float16 h16;
typedef __attribute__((ext_vector_type(16))) _Float16 v16h;
typedef __attribute__((ext_vector_type(8)))  _Float16 v8h;
typedef __attribute__((ext_vector_type(8)))  float    v8f;

// ---------------- WMMA fragment helpers ----------------
__device__ inline v16h frag2(const h16* p, int delta) {
  v8h lo = *(const v8h*)p;
  v8h hi = *(const v8h*)(p + delta);
  return __builtin_shufflevector(lo, hi, 0,1,2,3,4,5,6,7,8,9,10,11,12,13,14,15);
}
// A fragment: rows r0..r0+15, K = k0..k0+31, row-major with leading dim ld.
// lane<16: K 0..15; lane>=16: K 16..31   (matches fragB k order)
__device__ inline v16h fragA(const h16* A, int ld, int r0, int k0) {
  int lane = threadIdx.x & 31;
  return frag2(A + (size_t)(r0 + (lane & 15)) * ld + k0 + ((lane >> 4) << 4), 8);
}
// B fragment from N-major storage W[n][k]: cols n0..n0+15, K = k0..k0+31.
// lane<16: K 0..15 of col n; lane>=16: K 16..31           (contiguous k, same as fragA)
__device__ inline v16h fragB(const h16* W, int ld, int n0, int k0) {
  int lane = threadIdx.x & 31;
  return frag2(W + (size_t)(n0 + (lane & 15)) * ld + k0 + ((lane >> 4) << 4), 8);
}
// B fragment with per-lane (pre-clamped) row index r (used for attention k^T tiles)
__device__ inline v16h fragB_row(const h16* W, int ld, int r, int k0) {
  int lane = threadIdx.x & 31;
  return frag2(W + (size_t)r * ld + k0 + ((lane >> 4) << 4), 8);
}
// B fragment where K indexes sequence position s0 (16-aligned half-runs clamped)
__device__ inline v16h fragB_s(const h16* W, int ld, int n0, int s0) {
  int lane = threadIdx.x & 31;
  int ks = s0 + ((lane >> 4) << 4);
  if (ks < 0 || ks > Tt - 16) ks = 0;   // P==0 for these columns, data irrelevant
  return frag2(W + (size_t)(n0 + (lane & 15)) * ld + ks, 8);
}
__device__ inline v8f wmma_f16(v16h a, v16h b, v8f c) {
  return __builtin_amdgcn_wmma_f32_16x16x32_f16(false, a, false, b, (short)0, c, false, false);
}
__device__ inline float splus(float z) { return z > 20.f ? z : log1pf(expf(z)); }

// ---------------- small utility kernels ----------------
__global__ void k_f2h(const float* __restrict__ s, h16* __restrict__ d, int n) {
  int i = blockIdx.x * 256 + threadIdx.x;
  if (i < n) d[i] = (h16)s[i];
}
// x (D,T,B) -> h (b*T+t, d) fp32
__global__ void k_in(const float* __restrict__ x, float* __restrict__ h) {
  int i = blockIdx.x * 256 + threadIdx.x;
  if (i >= Dd * Tt * Bb) return;
  int b = i % Bb; int t = (i / Bb) % Tt; int d = i / (Bb * Tt);
  h[(size_t)(b * Tt + t) * Dd + d] = x[i];
}
// h (b*T+t, d) -> out (D,T,B)
__global__ void k_outc(const float* __restrict__ h, float* __restrict__ o) {
  int i = blockIdx.x * 256 + threadIdx.x;
  if (i >= Dd * Tt * Bb) return;
  int b = i % Bb; int t = (i / Bb) % Tt; int d = i / (Bb * Tt);
  o[i] = h[(size_t)(b * Tt + t) * Dd + d];
}
// per-block time conditioning: scale/shift (B,D) + gate (B)
__global__ void k_mod(const float* __restrict__ sw, const float* __restrict__ sb,
                      const float* __restrict__ hw, const float* __restrict__ hb,
                      const float* __restrict__ aw, const float* __restrict__ ab,
                      const float* __restrict__ te,
                      float* __restrict__ scale, float* __restrict__ shift,
                      float* __restrict__ gate) {
  int i = blockIdx.x * 256 + threadIdx.x;
  if (i < Bb * Dd) {
    int b = i / Dd, d = i % Dd;
    float as = sb[d], ah = hb[d];
    for (int kx = 0; kx < TDc; ++kx) {
      float tv = te[kx * Bb + b];
      as += sw[d * TDc + kx] * tv;
      ah += hw[d * TDc + kx] * tv;
    }
    scale[b * Dd + d] = as; shift[b * Dd + d] = ah;
  }
  if (i < Bb) {
    float a = ab[0];
    for (int kx = 0; kx < TDc; ++kx) a += aw[kx] * te[kx * Bb + i];
    gate[i] = 1.f / (1.f + expf(-a));
  }
}
// time projection for linear attention: tp[b][hd]
__global__ void k_tp(const float* __restrict__ tw, const float* __restrict__ tb,
                     const float* __restrict__ te, float* __restrict__ tp) {
  int i = blockIdx.x * 256 + threadIdx.x;
  if (i >= Bb * HD) return;
  int b = i / HD, j = i % HD;
  float a = tb[j];
  for (int kx = 0; kx < TDc; ++kx) a += tw[j * TDc + kx] * te[kx * Bb + b];
  tp[i] = a;
}
// time-conditioned layernorm over d: one wave per row, f16 out
__global__ void k_tcln(const float* __restrict__ h, const float* __restrict__ scale,
                       const float* __restrict__ shift, const float* __restrict__ g,
                       const float* __restrict__ bta, h16* __restrict__ out) {
  int wid = (blockIdx.x * blockDim.x + threadIdx.x) >> 5;
  if (wid >= Mm) return;
  int lane = threadIdx.x & 31;
  int b = wid / Tt;
  const float* row = h + (size_t)wid * Dd;
  float v[16]; float s = 0.f;
  #pragma unroll
  for (int j = 0; j < 16; ++j) { float x = row[lane + 32 * j]; v[j] = x; s += x; }
  for (int off = 16; off; off >>= 1) s += __shfl_xor(s, off, 32);
  float mean = s * (1.f / Dd);
  float ss = 0.f;
  #pragma unroll
  for (int j = 0; j < 16; ++j) { float d2 = v[j] - mean; ss += d2 * d2; }
  for (int off = 16; off; off >>= 1) ss += __shfl_xor(ss, off, 32);
  float inv = rsqrtf(ss * (1.f / Dd) + 1e-5f);
  #pragma unroll
  for (int j = 0; j < 16; ++j) {
    int d = lane + 32 * j;
    float nv = (v[j] - mean) * inv * g[d] + bta[d];
    nv = nv * (1.f + scale[b * Dd + d]) + shift[b * Dd + d];
    out[(size_t)wid * Dd + d] = (h16)nv;
  }
}

// ---------------- main WMMA GEMM: C = A(M,K) * W(N,K)^T ----------------
// MODE 0: f16 out (+bias, +tp[b][col&63]); 1: f32 out; 2: h += gate[b]*(..+bias); 3: f16 gelu(..+bias)
template <int MODE>
__global__ void k_gemm(const h16* __restrict__ A, const h16* __restrict__ W,
                       const float* __restrict__ bias, const float* __restrict__ tp,
                       void* __restrict__ out, float* __restrict__ hres,
                       const float* __restrict__ gate, int Mr, int N, int K) {
  int wave = threadIdx.x >> 5;
  int m0 = blockIdx.x * 16;
  int n0 = (blockIdx.y * (blockDim.x >> 5) + wave) * 16;
  if (n0 >= N || m0 >= Mr) return;
  v8f c = {};
  for (int k = 0; k < K; k += 32) {
    __builtin_prefetch(A + (size_t)(m0 + (threadIdx.x & 15)) * K + k + 64, 0, 1);
    c = wmma_f16(fragA(A, K, m0, k), fragB(W, K, n0, k), c);
  }
  int lane = threadIdx.x & 31;
  int col = n0 + (lane & 15);
  int rbase = m0 + ((lane >> 4) << 3);
  #pragma unroll
  for (int r = 0; r < 8; ++r) {
    int row = rbase + r;
    float v = c[r];
    if (bias) v += bias[col];
    if (MODE == 0) {
      if (tp) v += tp[(row / Tt) * HD + (col & (HD - 1))];
      ((h16*)out)[(size_t)row * N + col] = (h16)v;
    } else if (MODE == 1) {
      ((float*)out)[(size_t)row * N + col] = v;
    } else if (MODE == 2) {
      int b = row / Tt;
      hres[(size_t)row * Dd + col] += gate[b] * v;
    } else {
      float gl = 0.5f * v * (1.f + erff(v * 0.70710678118f));
      ((h16*)out)[(size_t)row * N + col] = (h16)gl;
    }
  }
}

// ---------------- dlinoss sequential scan ----------------
__global__ void k_scan(const float* __restrict__ logdt, const float* __restrict__ logk,
                       const float* __restrict__ logc, const float* __restrict__ u,
                       h16* __restrict__ seq) {
  int i = blockIdx.x * 256 + threadIdx.x;
  if (i >= Bb * Ss) return;
  int b = i / Ss, s = i % Ss;
  float dt = expf(logdt[s]), kk = expf(logk[s]), cc = expf(logc[s]);
  float f = 1.f / (1.f + dt * cc);
  float vr = f, sc = -dt * kk * f, ig = dt * f;
  float vel = 0.f, pos = 0.f;
  for (int t = 0; t < Tt; ++t) {
    float ut = u[(size_t)(b * Tt + t) * Ss + s];
    vel = vr * vel + sc * pos + ig * ut;
    pos += dt * vel;
    seq[(size_t)(b * Tt + t) * Ss + s] = (h16)pos;
  }
}

// ---------------- phi: LN(softplus(f*pos)) for cos & sin ----------------
__device__ inline void phi_one(float x0, float x1, float p0, float p1,
                               const float* fg, const float* fb, int lane,
                               h16* out, size_t i0, size_t i1) {
  float y0 = splus(x0 * p0), y1 = splus(x1 * p1);
  float s = y0 + y1;
  for (int off = 1; off < 32; off <<= 1) s += __shfl_xor(s, off, 32);
  float mean = s * (1.f / 64.f);
  float d0 = y0 - mean, d1 = y1 - mean;
  float vv = d0 * d0 + d1 * d1;
  for (int off = 1; off < 32; off <<= 1) vv += __shfl_xor(vv, off, 32);
  float inv = rsqrtf(vv * (1.f / 64.f) + 1e-5f);
  out[i0] = (h16)(d0 * inv * fg[lane] + fb[lane]);
  out[i1] = (h16)(d1 * inv * fg[lane + 32] + fb[lane + 32]);
}
__global__ void k_phi(const h16* __restrict__ f, const float* __restrict__ pc,
                      const float* __restrict__ ps, const float* __restrict__ fg,
                      const float* __restrict__ fb, h16* __restrict__ oc,
                      h16* __restrict__ os, int tr) {
  int wid = (blockIdx.x * blockDim.x + threadIdx.x) >> 5;
  if (wid >= MH) return;
  int lane = threadIdx.x & 31;
  int hh = wid % Hh; int bt = wid / Hh; int t = bt % Tt; int b = bt / Tt;
  float x0 = (float)f[(size_t)wid * HD + lane];
  float x1 = (float)f[(size_t)wid * HD + lane + 32];
  size_t i0, i1;
  if (tr) { // e-major over t (for k features): [((b*H+h)*64+e)*T + t]
    size_t base = ((size_t)(b * Hh + hh) * HD) * Tt + t;
    i0 = base + (size_t)lane * Tt; i1 = base + (size_t)(lane + 32) * Tt;
  } else {  // natural (for q features)
    i0 = (size_t)wid * HD + lane; i1 = i0 + 32;
  }
  phi_one(x0, x1, pc[t * HD + lane], pc[t * HD + lane + 32], fg, fb, lane, oc, i0, i1);
  phi_one(x0, x1, ps[t * HD + lane], ps[t * HD + lane + 32], fg, fb, lane, os, i0, i1);
}

// v (b*T+t, h*64+d) -> vt[((b*H+h)*64+d)*T + t]
__global__ void k_vt(const h16* __restrict__ v, h16* __restrict__ vt) {
  size_t i = (size_t)blockIdx.x * 256 + threadIdx.x;
  if (i >= (size_t)Mm * Dd) return;
  int col = (int)(i % Dd); size_t row = i / Dd;
  int hh = col >> 6, d = col & 63; int t = (int)(row % Tt); int b = (int)(row / Tt);
  vt[((size_t)(b * Hh + hh) * HD + d) * Tt + t] = v[i];
}

// ctx[b,h,d,e] = sum_t v[t,d]*kc[t,e]  (batched 64x64xT WMMA, cos & sin share A)
__global__ void k_ctx(const h16* __restrict__ vt, const h16* __restrict__ kct,
                      const h16* __restrict__ kst, h16* __restrict__ cc,
                      h16* __restrict__ cs) {
  int bh = blockIdx.x, dt = blockIdx.y, et = threadIdx.x >> 5;
  const h16* Ab = vt + (size_t)bh * HD * Tt;
  const h16* Bc = kct + (size_t)bh * HD * Tt;
  const h16* Bs = kst + (size_t)bh * HD * Tt;
  v8f c0 = {}, c1 = {};
  for (int k = 0; k < Tt; k += 32) {
    v16h a = fragA(Ab, Tt, dt * 16, k);
    c0 = wmma_f16(a, fragB(Bc, Tt, et * 16, k), c0);
    c1 = wmma_f16(a, fragB(Bs, Tt, et * 16, k), c1);
  }
  int lane = threadIdx.x & 31;
  int col = et * 16 + (lane & 15);
  int rb = dt * 16 + ((lane >> 4) << 3);
  #pragma unroll
  for (int r = 0; r < 8; ++r) {
    cc[(size_t)bh * 4096 + (rb + r) * 64 + col] = (h16)c0[r];
    cs[(size_t)bh * 4096 + (rb + r) * 64 + col] = (h16)c1[r];
  }
}

// out[b,t,h,d] = sum_e qc[t,e]*ctx_c[d,e] + qs[t,e]*ctx_s[d,e]
__global__ void k_outla(const h16* __restrict__ qc, const h16* __restrict__ qs,
                        const h16* __restrict__ cc, const h16* __restrict__ cs,
                        h16* __restrict__ att) {
  int bh = blockIdx.x, b = bh / Hh, hh = bh % Hh;
  int t0 = blockIdx.y * 16, dt = threadIdx.x >> 5;
  const h16* Ac = qc + ((size_t)(b * Tt) * Hh + hh) * HD;  // row stride 512
  const h16* As = qs + ((size_t)(b * Tt) * Hh + hh) * HD;
  const h16* Bc = cc + (size_t)bh * 4096;                  // N-major [d][e], ld 64
  const h16* Bs = cs + (size_t)bh * 4096;
  v8f c = {};
  for (int k = 0; k < HD; k += 32) {
    c = wmma_f16(fragA(Ac, Hh * HD, t0, k), fragB(Bc, HD, dt * 16, k), c);
    c = wmma_f16(fragA(As, Hh * HD, t0, k), fragB(Bs, HD, dt * 16, k), c);
  }
  int lane = threadIdx.x & 31;
  int col = dt * 16 + (lane & 15);
  int rb = t0 + ((lane >> 4) << 3);
  #pragma unroll
  for (int r = 0; r < 8; ++r)
    att[((size_t)b * Tt + rb + r) * Dd + hh * HD + col] = (h16)c[r];
}

// ---------------- banded (|t-s|<=WIN) attention, flash-style ----------------
__global__ void k_swattn(const h16* __restrict__ q, const h16* __restrict__ kk,
                         const h16* __restrict__ vt, h16* __restrict__ att) {
  __shared__ __align__(16) h16 Pl[4][16 * 32];
  int w = threadIdx.x >> 5, lane = threadIdx.x & 31;
  int bh = blockIdx.x, b = bh / Hh, hh = bh % Hh;
  int t0 = (blockIdx.y * 4 + w) * 16;
  const h16* qb = q + ((size_t)b * Tt) * Dd + hh * HD;
  const h16* kb = kk + ((size_t)b * Tt) * Dd + hh * HD;
  const h16* vb = vt + (size_t)bh * HD * Tt;
  h16* P = Pl[w];
  v16h aq0 = fragA(qb, Dd, t0, 0);
  v16h aq1 = fragA(qb, Dd, t0, 32);
  v8f o0 = {}, o1 = {}, o2 = {}, o3 = {};
  float mrow[8], lrow[8];
  #pragma unroll
  for (int r = 0; r < 8; ++r) { mrow[r] = -1e30f; lrow[r] = 0.f; }
  int rofs = ((lane >> 4) << 3);
  for (int j = 0; j < 17; ++j) {
    int s0 = t0 - WIN + j * 32;
    int rA = s0 + (lane & 15), rB = rA + 16;
    int rAc = min(max(rA, 0), Tt - 1), rBc = min(max(rB, 0), Tt - 1);
    v8f sA = {}, sB = {};
    sA = wmma_f16(aq0, fragB_row(kb, Dd, rAc, 0), sA);
    sA = wmma_f16(aq1, fragB_row(kb, Dd, rAc, 32), sA);
    sB = wmma_f16(aq0, fragB_row(kb, Dd, rBc, 0), sB);
    sB = wmma_f16(aq1, fragB_row(kb, Dd, rBc, 32), sB);
    float txA[8], txB[8], nm[8];
    #pragma unroll
    for (int r = 0; r < 8; ++r) {
      int row = t0 + rofs + r;
      float xa = sA[r] * 0.125f, xb = sB[r] * 0.125f;
      float ta = xa - splus(-xa), tb = xb - splus(-xb);  // x + log_sigmoid(x)
      int da = row - rA; if (da < 0) da = -da;
      int db = row - rB; if (db < 0) db = -db;
      txA[r] = (rA >= 0 && rA < Tt && da <= WIN) ? ta : -1e30f;
      txB[r] = (rB >= 0 && rB < Tt && db <= WIN) ? tb : -1e30f;
      nm[r] = fmaxf(txA[r], txB[r]);
    }
    for (int off = 1; off < 16; off <<= 1)
      #pragma unroll
      for (int r = 0; r < 8; ++r) nm[r] = fmaxf(nm[r], __shfl_xor(nm[r], off, 32));
    float rs[8];
    #pragma unroll
    for (int r = 0; r < 8; ++r) {
      float mn = fmaxf(mrow[r], nm[r]);
      float sf = expf(mrow[r] - mn);
      mrow[r] = mn;
      lrow[r] *= sf;
      o0[r] *= sf; o1[r] *= sf; o2[r] *= sf; o3[r] *= sf;
      float pA = txA[r] <= -1e29f ? 0.f : expf(txA[r] - mn);
      float pB = txB[r] <= -1e29f ? 0.f : expf(txB[r] - mn);
      P[(rofs + r) * 32 + (lane & 15)] = (h16)pA;
      P[(rofs + r) * 32 + 16 + (lane & 15)] = (h16)pB;
      rs[r] = pA + pB;
    }
    for (int off = 1; off < 16; off <<= 1)
      #pragma unroll
      for (int r = 0; r < 8; ++r) rs[r] += __shfl_xor(rs[r], off, 32);
    #pragma unroll
    for (int r = 0; r < 8; ++r) lrow[r] += rs[r];
    __syncthreads();
    v16h ap = fragA(P, 32, 0, 0);
    o0 = wmma_f16(ap, fragB_s(vb, Tt, 0, s0), o0);
    o1 = wmma_f16(ap, fragB_s(vb, Tt, 16, s0), o1);
    o2 = wmma_f16(ap, fragB_s(vb, Tt, 32, s0), o2);
    o3 = wmma_f16(ap, fragB_s(vb, Tt, 48, s0), o3);
    __syncthreads();
  }
  #pragma unroll
  for (int r = 0; r < 8; ++r) {
    float inv = 1.f / lrow[r];
    int row = t0 + rofs + r;
    size_t base = ((size_t)b * Tt + row) * Dd + hh * HD + (lane & 15);
    att[base + 0]  = (h16)(o0[r] * inv);
    att[base + 16] = (h16)(o1[r] * inv);
    att[base + 32] = (h16)(o2[r] * inv);
    att[base + 48] = (h16)(o3[r] * inv);
  }
}

// ---------------- swiglu gate ----------------
__global__ void k_swiglu(const h16* __restrict__ e, h16* __restrict__ g) {
  size_t i = (size_t)blockIdx.x * 256 + threadIdx.x;
  if (i >= (size_t)Mm * (HID / 2)) return;
  int col = (int)(i % (HID / 2)); size_t row = i / (HID / 2);
  float a = (float)e[row * HID + col];
  float bv = (float)e[row * HID + HID / 2 + col];
  float s = a / (1.f + expf(-a));
  g[i] = (h16)(s * bv);
}

// ---------------- host ----------------
// Input index map (jax pytree: dict keys sorted). See analysis for derivation.
extern "C" void kernel_launch(void* const* d_in, const int* in_sizes, int n_in,
                              void* d_out, int out_size, void* d_ws, size_t ws_size,
                              hipStream_t stream) {
  (void)in_sizes; (void)n_in; (void)out_size; (void)ws_size;
  char* ws = (char*)d_ws;
  size_t off = 0;
  auto take = [&](size_t bytes) -> char* {
    char* p = ws + off; off += (bytes + 255) & ~(size_t)255; return p;
  };
  float* hbuf = (float*)take((size_t)Mm * Dd * 4);   // residual stream
  h16*   n16  = (h16*)take((size_t)Mm * Dd * 2);     // normalized activations
  char* slot[8];
  for (int i = 0; i < 8; ++i) slot[i] = take((size_t)8 << 20);  // 8 x 8MB pool

  auto IN = [&](int i) { return (const float*)d_in[i]; };
  auto conv = [&](int idx, size_t n) -> h16* {
    h16* p = (h16*)take(n * 2);
    k_f2h<<<dim3((unsigned)((n + 255) / 256)), dim3(256), 0, stream>>>(IN(idx), p, (int)n);
    return p;
  };
  // weights -> f16 (N-major as stored: w is (fout, fin))
  h16 *Wq_la = conv(17, 512 * 512), *Wk_la = conv(7, 512 * 512);
  h16 *Wv_la = conv(25, 512 * 512), *Wo_la = conv(13, 512 * 512);
  h16 *Wqf1 = conv(19, 128 * 64), *Wqf2 = conv(21, 64 * 128);
  h16 *Wkf1 = conv(9, 128 * 64),  *Wkf2 = conv(11, 64 * 128);
  h16 *Win = conv(58, 256 * 512), *Wout = conv(62, 512 * 256);
  h16 *Wq_sw = conv(68, 512 * 512), *Wk_sw = conv(64, 512 * 512);
  h16 *Wv_sw = conv(70, 512 * 512), *Wo_sw = conv(66, 512 * 512);
  h16 *Wexp = conv(3, 768 * 512), *Wcon = conv(1, 512 * 384);
  float* scale = (float*)take(Bb * Dd * 4);
  float* shift = (float*)take(Bb * Dd * 4);
  float* gate  = (float*)take(64 * 4);
  float* tp    = (float*)take(Bb * HD * 4);
  h16* ctxc = (h16*)take((size_t)64 * 4096 * 2);
  h16* ctxs = (h16*)take((size_t)64 * 4096 * 2);

  auto g16 = [&](const h16* A, const h16* W, const float* bias, const float* tpv,
                 h16* out, int Mr, int N, int K) {
    k_gemm<0><<<dim3(Mr / 16, (N + 63) / 64), dim3(128), 0, stream>>>(
        A, W, bias, tpv, (void*)out, nullptr, nullptr, Mr, N, K);
  };
  auto g32 = [&](const h16* A, const h16* W, float* out, int Mr, int N, int K) {
    k_gemm<1><<<dim3(Mr / 16, (N + 63) / 64), dim3(128), 0, stream>>>(
        A, W, nullptr, nullptr, (void*)out, nullptr, nullptr, Mr, N, K);
  };
  auto gres = [&](const h16* A, const h16* W, const float* bias, int N, int K) {
    k_gemm<2><<<dim3(Mm / 16, (N + 63) / 64), dim3(128), 0, stream>>>(
        A, W, bias, nullptr, nullptr, hbuf, gate, Mm, N, K);
  };
  auto ggelu = [&](const h16* A, const h16* W, const float* bias, h16* out,
                   int Mr, int N, int K) {
    k_gemm<3><<<dim3(Mr / 16, (N + 63) / 64), dim3(128), 0, stream>>>(
        A, W, bias, nullptr, (void*)out, nullptr, nullptr, Mr, N, K);
  };
  auto tcln = [&](int L) {  // L = base index of ln params (ab.b,ab.w,b,g,scale.b,scale.w,shift.b,shift.w)
    k_mod<<<dim3(16), dim3(256), 0, stream>>>(IN(L + 5), IN(L + 4), IN(L + 7), IN(L + 6),
                                              IN(L + 1), IN(L + 0), IN(71),
                                              scale, shift, gate);
    k_tcln<<<dim3(Mm / 8), dim3(256), 0, stream>>>(hbuf, scale, shift, IN(L + 3), IN(L + 2), n16);
  };

  // residual stream = x
  k_in<<<dim3(Dd * Tt * Bb / 256), dim3(256), 0, stream>>>(IN(72), hbuf);

  // ---- block 1: dlinoss ----
  tcln(26);
  float* u = (float*)slot[0];
  g32(n16, Win, u, Mm, Ss, Dd);
  h16* seq = (h16*)slot[1];
  k_scan<<<dim3(Bb * Ss / 256), dim3(256), 0, stream>>>(IN(60), IN(61), IN(59), u, seq);
  gres(seq, Wout, nullptr, Dd, Ss);

  // ---- block 2: linear attention ----
  tcln(34);
  k_tp<<<dim3(2), dim3(256), 0, stream>>>(IN(23), IN(22), IN(71), tp);
  h16 *qbuf = (h16*)slot[0], *kbuf = (h16*)slot[1], *vbuf = (h16*)slot[2];
  g16(n16, Wq_la, IN(16), tp, qbuf, Mm, Dd, Dd);
  g16(n16, Wk_la, IN(6), tp, kbuf, Mm, Dd, Dd);
  g16(n16, Wv_la, IN(24), nullptr, vbuf, Mm, Dd, Dd);
  h16* g1 = (h16*)slot[3];                    // 16MB, spans slot[3..4]
  ggelu(qbuf, Wqf1, IN(18), g1, MH, 128, HD);
  h16* qf = (h16*)slot[5];
  g16(g1, Wqf2, IN(20), nullptr, qf, MH, HD, 128);
  ggelu(kbuf, Wkf1, IN(8), g1, MH, 128, HD);
  h16* kf = (h16*)slot[6];
  g16(g1, Wkf2, IN(10), nullptr, kf, MH, HD, 128);
  h16 *qc = (h16*)slot[0], *qs = (h16*)slot[1];           // q,k dead
  k_phi<<<dim3(MH / 8), dim3(256), 0, stream>>>(qf, IN(14), IN(15), IN(5), IN(4), qc, qs, 0);
  h16 *kct = (h16*)slot[3], *kst = (h16*)slot[4];          // g1 dead
  k_phi<<<dim3(MH / 8), dim3(256), 0, stream>>>(kf, IN(14), IN(15), IN(5), IN(4), kct, kst, 1);
  h16* vt = (h16*)slot[7];
  k_vt<<<dim3(Mm * Dd / 256), dim3(256), 0, stream>>>(vbuf, vt);
  k_ctx<<<dim3(Bb * Hh, 4), dim3(128), 0, stream>>>(vt, kct, kst, ctxc, ctxs);
  h16* att = (h16*)slot[2];                                // v dead
  k_outla<<<dim3(Bb * Hh, Tt / 16), dim3(128), 0, stream>>>(qc, qs, ctxc, ctxs, att);
  gres(att, Wo_la, IN(12), Dd, Dd);

  // ---- block 3: sliding-window attention ----
  tcln(42);
  g16(n16, Wq_sw, IN(67), nullptr, qbuf, Mm, Dd, Dd);
  g16(n16, Wk_sw, IN(63), nullptr, kbuf, Mm, Dd, Dd);
  g16(n16, Wv_sw, IN(69), nullptr, vbuf, Mm, Dd, Dd);
  k_vt<<<dim3(Mm * Dd / 256), dim3(256), 0, stream>>>(vbuf, vt);
  h16* att3 = (h16*)slot[3];
  k_swattn<<<dim3(Bb * Hh, Tt / 64), dim3(128), 0, stream>>>(qbuf, kbuf, vt, att3);
  gres(att3, Wo_sw, IN(65), Dd, Dd);

  // ---- block 4: swiglu ----
  tcln(50);
  h16* e = (h16*)slot[0];                                  // 12MB, spans slot[0..1]
  g16(n16, Wexp, IN(2), nullptr, e, Mm, HID, Dd);
  h16* gsw = (h16*)slot[2];
  k_swiglu<<<dim3(Mm * (HID / 2) / 256), dim3(256), 0, stream>>>(e, gsw);
  gres(gsw, Wcon, IN(0), Dd, HID / 2);

  k_outc<<<dim3(Dd * Tt * Bb / 256), dim3(256), 0, stream>>>(hbuf, (float*)d_out);
}
